// ChebyNet_74569222193257
// MI455X (gfx1250) — matrix operand order repaired and checked by reading, not compiled, not run on hardware
//
#include <hip/hip_runtime.h>
#include <hip/hip_bf16.h>

#define RC_    6.0f
#define RMIN_  0.5f
#define BETA_  12
#define M1_    24
#define M2_    6
#define B_     4
#define N_     1024
#define NT_    2
#define MN_    128
#define K_     256
#define NFEAT_ 144
#define PI_    3.14159265358979f

typedef float  f32x8   __attribute__((ext_vector_type(8)));
typedef __bf16 bf16x16 __attribute__((ext_vector_type(16)));
typedef __bf16 bf16x8  __attribute__((ext_vector_type(8)));

// ---------------------------------------------------------------------------
// WMMA helpers (CDNA5 v_wmma_f32_16x16x32_bf16, wave32)
// ---------------------------------------------------------------------------
__device__ inline f32x8 wmma_bf16(bf16x16 a, bf16x16 b, f32x8 c) {
  return __builtin_amdgcn_wmma_f32_16x16x32_bf16(false, a, false, b, (short)0, c,
                                                 false, false);
}

// A matrix 16x32 bf16 from a bf16 LDS tile (row-major, given element stride).
// ISA layout: lane M = lane&15, half = lane>>4;
// element e -> K = (e%8) + 16*(e/8) + 8*half  => two contiguous 8-element runs.
__device__ inline bf16x16 frag_A_lds(const __bf16* sA, int stride, int lane,
                                     int kbase) {
  int M = lane & 15, half = lane >> 4;
  const __bf16* p = sA + M * stride + kbase + (half << 3);
  bf16x8 lo = *(const bf16x8*)p;        // K = kbase+8h .. +7
  bf16x8 hi = *(const bf16x8*)(p + 16); // K = kbase+16+8h .. +7
  return __builtin_shufflevector(lo, hi, 0, 1, 2, 3, 4, 5, 6, 7, 8, 9, 10, 11, 12,
                                 13, 14, 15);
}

// B matrix 32x16 bf16 from a K-contiguous bf16 weight layout W[col][K] (ld = K dim).
// ISA layout: lane col = lane&15; half = lane>>4; element e -> K = kbase + (e%8) + 16*(e/8) + 8*half
// => two aligned 16-byte contiguous loads.
__device__ inline bf16x16 frag_B_pack(const __bf16* W, int ld, int col, int lane,
                                      int kbase) {
  const __bf16* p = W + (size_t)col * ld + kbase + ((lane >> 4) << 3);
  bf16x8 lo = *(const bf16x8*)p;
  bf16x8 hi = *(const bf16x8*)(p + 16);
  return __builtin_shufflevector(lo, hi, 0, 1, 2, 3, 4, 5, 6, 7, 8, 9, 10, 11, 12,
                                 13, 14, 15);
}

// ---------------------------------------------------------------------------
// Kernel 0: zero the output buffer (Etot/Force/Virial are accumulated atomically)
// ---------------------------------------------------------------------------
__global__ void zero_out_k(float* p, int n) {
  int i = blockIdx.x * blockDim.x + threadIdx.x;
  if (i < n) p[i] = 0.0f;
}

// ---------------------------------------------------------------------------
// Kernel 0b: one-time bf16 weight conversion into K-contiguous layouts.
//  W0T [NT][128][160] : W0T[t][h][f] = W0[t][f][h], zero padded f in [144,160)
//  W1T [NT][128][128] : W1T[t][g][h] = W1[t][h][g]
//  W0b [NT][144][128] : bf16 copy of W0   (K-contiguous for dfeat = dz0 x W0^T)
//  W1b [NT][128][128] : bf16 copy of W1   (K-contiguous for dh = dz1 x W1^T)
// ---------------------------------------------------------------------------
#define N_W0T (NT_ * 128 * 160)
#define N_W1T (NT_ * 128 * 128)
#define N_W0B (NT_ * NFEAT_ * 128)
#define N_W1B (NT_ * 128 * 128)

__global__ void cvt_weights_k(const float* __restrict__ W0,
                              const float* __restrict__ W1, __bf16* __restrict__ W0T,
                              __bf16* __restrict__ W1T, __bf16* __restrict__ W0b,
                              __bf16* __restrict__ W1b) {
  int i = blockIdx.x * blockDim.x + threadIdx.x;
  if (i < N_W0T) {
    int t = i / (128 * 160), h = (i / 160) & 127, f = i % 160;
    W0T[i] = (__bf16)((f < NFEAT_) ? W0[(size_t)t * NFEAT_ * 128 + f * 128 + h]
                                   : 0.0f);
    return;
  }
  int j = i - N_W0T;
  if (j < N_W1T) {
    int t = j / (128 * 128), g = (j / 128) & 127, h = j & 127;
    W1T[j] = (__bf16)W1[(size_t)t * 128 * 128 + h * 128 + g];
    return;
  }
  int k2 = j - N_W1T;
  if (k2 < N_W0B) {
    W0b[k2] = (__bf16)W0[k2];
    return;
  }
  int k3 = k2 - N_W0B;
  if (k3 < N_W1B) W1b[k3] = (__bf16)W1[k3];
}

// ---------------------------------------------------------------------------
// Kernel 1: descriptor forward. One wave32 per atom.
// Writes R (24x4, already /MN) and feat (144, already *scale) to workspace.
// ---------------------------------------------------------------------------
__global__ __launch_bounds__(32) void feat_fwd_k(
    const float* __restrict__ ImageDR, const int* __restrict__ list_neigh,
    const int* __restrict__ ti_map, const float* __restrict__ c_param,
    const float* __restrict__ scale, float* __restrict__ wsFeat,
    float* __restrict__ wsR) {
  int atom = blockIdx.x;
  int n = atom % N_;
  int lane = threadIdx.x;

  __shared__ float sC[NT_ * M1_ * BETA_];  // c_param[ti][tk][f][beta]
  __shared__ float sRed[32][96];
  __shared__ float sR[96];

  int ti = ti_map[n];
  for (int i = lane; i < NT_ * M1_ * BETA_; i += 32)
    sC[i] = c_param[(size_t)ti * NT_ * M1_ * BETA_ + i];
  __syncthreads();

  float Racc[M1_][4];
#pragma unroll
  for (int f = 0; f < M1_; ++f)
#pragma unroll
    for (int c = 0; c < 4; ++c) Racc[f][c] = 0.0f;

  const float invspan = 1.0f / (RC_ - RMIN_);
  for (int it = 0; it < K_ / 32; ++it) {
    int k = it * 32 + lane;
    const float* p = ImageDR + ((size_t)atom * K_ + k) * 4;
    float x = p[1], y = p[2], z = p[3];
    int ln = list_neigh[(size_t)atom * K_ + k];
    float valid = (ln >= 0) ? 1.0f : 0.0f;
    float r = sqrtf(x * x + y * y + z * z + 1e-12f);
    float u = fminf(fmaxf((r - RMIN_) * invspan, 0.0f), 1.0f);
    float fc = 0.5f * (cosf(PI_ * u) + 1.0f) * valid;
    float xc = 2.0f * (r - RMIN_) * invspan - 1.0f;
    float T[BETA_];
    T[0] = 1.0f;
    T[1] = xc;
#pragma unroll
    for (int j = 2; j < BETA_; ++j) T[j] = 2.0f * xc * T[j - 1] - T[j - 2];
    float invr = valid / r;
    float sr1 = x * invr, sr2 = y * invr, sr3 = z * invr;
    int tk = k >> 7;  // k / MN_
    const float* cc = &sC[tk * M1_ * BETA_];
#pragma unroll
    for (int f = 0; f < M1_; ++f) {
      float sv = 0.0f;
#pragma unroll
      for (int j = 0; j < BETA_; ++j) sv = fmaf(cc[f * BETA_ + j], T[j], sv);
      sv *= fc;
      Racc[f][0] += sv * valid;
      Racc[f][1] += sv * sr1;
      Racc[f][2] += sv * sr2;
      Racc[f][3] += sv * sr3;
    }
  }

#pragma unroll
  for (int f = 0; f < M1_; ++f)
#pragma unroll
    for (int c = 0; c < 4; ++c) sRed[lane][f * 4 + c] = Racc[f][c];
  __syncthreads();

  for (int q = 0; q < 3; ++q) {
    int j = lane + q * 32;
    float s = 0.0f;
    for (int l = 0; l < 32; ++l) s += sRed[l][j];
    s *= (1.0f / MN_);
    sR[j] = s;
    wsR[(size_t)atom * 96 + j] = s;
  }
  __syncthreads();

  for (int idx = lane; idx < NFEAT_; idx += 32) {
    int f = idx / M2_, g = idx % M2_;
    float acc = 0.0f;
#pragma unroll
    for (int c = 0; c < 4; ++c) acc += sR[f * 4 + c] * sR[g * 4 + c];
    wsFeat[(size_t)atom * NFEAT_ + idx] = acc * scale[idx];
  }
}

// ---------------------------------------------------------------------------
// Kernel 2: MLP forward + backward with WMMA bf16 (f32 accumulate).
// 16-atom tiles; both atom types evaluated, selected by ti_map.
// All GEMM operands staged as bf16 (LDS tiles + pre-converted weights).
// ---------------------------------------------------------------------------
__global__ __launch_bounds__(256) void mlp_wmma_k(
    const float* __restrict__ wsFeat, float* __restrict__ wsDfeat,
    const int* __restrict__ ti_map, const __bf16* __restrict__ W0T,
    const __bf16* __restrict__ W1T, const __bf16* __restrict__ W0b,
    const __bf16* __restrict__ W1b, const float* __restrict__ b0,
    const float* __restrict__ b1, const float* __restrict__ W2,
    const float* __restrict__ b2, const float* __restrict__ ener_shift,
    float* __restrict__ outEtot, float* __restrict__ outEi) {
  __shared__ __bf16 sFeatB[16 * 160];  // feat, K padded 144 -> 160 with zeros
  __shared__ __bf16 sHb[16 * 128];     // h   (GEMM A operand)
  __shared__ __bf16 sXb[16 * 128];     // dz1 then dz0 (GEMM A operand)
  __shared__ float sE[16];
  __shared__ int sType[16];

  int tid = threadIdx.x, lane = tid & 31, wave = tid >> 5;
  int tileBase = blockIdx.x * 16;  // flat atom index base (tiles never cross b)
  int bIdx = tileBase / N_;

  for (int idx = tid; idx < 16 * 160; idx += 256) {
    int row = idx / 160, col = idx % 160;
    sFeatB[idx] = (__bf16)(
        (col < NFEAT_) ? wsFeat[(size_t)(tileBase + row) * NFEAT_ + col] : 0.0f);
  }
  if (tid < 16) sType[tid] = ti_map[(tileBase + tid) % N_];
  __syncthreads();

  int laneM = lane & 15, half = lane >> 4;
  int col = wave * 16 + laneM;  // column this lane owns for the 128-wide layers

  for (int t = 0; t < NT_; ++t) {
    __syncthreads();
    if (tid < 16) sE[tid] = 0.0f;
    __syncthreads();

    const __bf16* W0Tt = W0T + (size_t)t * 128 * 160;
    const __bf16* W1Tt = W1T + (size_t)t * 128 * 128;
    const __bf16* W0bt = W0b + (size_t)t * NFEAT_ * 128;
    const __bf16* W1bt = W1b + (size_t)t * 128 * 128;
    const float* b0t = b0 + t * 128;
    const float* b1t = b1 + t * 128;
    const float* W2t = W2 + t * 128;

    // ---- z0 = feat x W0 ; h = tanh(z0 + b0) ----
    f32x8 acc = {0, 0, 0, 0, 0, 0, 0, 0};
#pragma unroll
    for (int s = 0; s < 5; ++s)
      acc = wmma_bf16(frag_A_lds(sFeatB, 160, lane, s * 32),
                      frag_B_pack(W0Tt, 160, col, lane, s * 32), acc);
    float bb0 = b0t[col];
    float hreg[8];
#pragma unroll
    for (int v = 0; v < 8; ++v) {
      int row = v + 8 * half;  // f32 C layout: VGPR v -> M = v (+8 for upper half)
      hreg[v] = tanhf(acc[v] + bb0);
      sHb[row * 128 + col] = (__bf16)hreg[v];
    }
    __syncthreads();

    // ---- z1 = h x W1 ; t1 = tanh(z1 + b1) ; h2 = t1 + h ; E += h2 . W2 ----
    f32x8 acc1 = {0, 0, 0, 0, 0, 0, 0, 0};
#pragma unroll
    for (int s = 0; s < 4; ++s)
      acc1 = wmma_bf16(frag_A_lds(sHb, 128, lane, s * 32),
                       frag_B_pack(W1Tt, 128, col, lane, s * 32), acc1);
    float bb1 = b1t[col], w2c = W2t[col];
#pragma unroll
    for (int v = 0; v < 8; ++v) {
      int row = v + 8 * half;
      float t1 = tanhf(acc1[v] + bb1);
      atomicAdd(&sE[row], (t1 + hreg[v]) * w2c);
      sXb[row * 128 + col] = (__bf16)(w2c * (1.0f - t1 * t1));  // dE/dz1
    }
    __syncthreads();

    // ---- dh = dz1 x W1^T + W2 (residual) ; dz0 = dh * (1 - h^2) ----
    f32x8 accD = {0, 0, 0, 0, 0, 0, 0, 0};
#pragma unroll
    for (int s = 0; s < 4; ++s)
      accD = wmma_bf16(frag_A_lds(sXb, 128, lane, s * 32),
                       frag_B_pack(W1bt, 128, col, lane, s * 32), accD);
    float dz0reg[8];
#pragma unroll
    for (int v = 0; v < 8; ++v)
      dz0reg[v] = (accD[v] + w2c) * (1.0f - hreg[v] * hreg[v]);
    __syncthreads();
#pragma unroll
    for (int v = 0; v < 8; ++v) sXb[(v + 8 * half) * 128 + col] = (__bf16)dz0reg[v];
    __syncthreads();

    // ---- dfeat = dz0 x W0^T (144 cols -> 9 tiles over 8 waves) ----
    for (int ct = wave; ct < 9; ct += 8) {
      int colF = ct * 16 + laneM;
      f32x8 accF = {0, 0, 0, 0, 0, 0, 0, 0};
#pragma unroll
      for (int s = 0; s < 4; ++s)
        accF = wmma_bf16(frag_A_lds(sXb, 128, lane, s * 32),
                         frag_B_pack(W0bt, 128, colF, lane, s * 32), accF);
#pragma unroll
      for (int v = 0; v < 8; ++v) {
        int row = v + 8 * half;
        if (sType[row] == t)
          wsDfeat[(size_t)(tileBase + row) * NFEAT_ + colF] = accF[v];
      }
    }
    __syncthreads();

    if (tid < 16 && sType[tid] == t) {
      float Ei = sE[tid] + b2[t] + ener_shift[t];
      outEi[tileBase + tid] = Ei;
      atomicAdd(&outEtot[bIdx], Ei);
    }
  }
}

// ---------------------------------------------------------------------------
// Kernel 3: descriptor backward. One wave32 per atom.
// Chain rule dfeat -> dR -> per-neighbor g; scatter forces; virial.
// ---------------------------------------------------------------------------
__global__ __launch_bounds__(32) void feat_bwd_k(
    const float* __restrict__ ImageDR, const int* __restrict__ list_neigh,
    const int* __restrict__ ti_map, const float* __restrict__ c_param,
    const float* __restrict__ scale, const float* __restrict__ wsR,
    const float* __restrict__ wsDfeat, float* __restrict__ outForce,
    float* __restrict__ outVirial) {
  int atom = blockIdx.x;
  int b = atom / N_;
  int n = atom % N_;
  int lane = threadIdx.x;

  __shared__ float sC[NT_ * M1_ * BETA_];
  __shared__ float sDF[NFEAT_];
  __shared__ float sR[96];
  __shared__ float sDR[96];

  int ti = ti_map[n];
  for (int i = lane; i < NT_ * M1_ * BETA_; i += 32)
    sC[i] = c_param[(size_t)ti * NT_ * M1_ * BETA_ + i];
  for (int i = lane; i < NFEAT_; i += 32)
    sDF[i] = wsDfeat[(size_t)atom * NFEAT_ + i] * scale[i];
  for (int i = lane; i < 96; i += 32) sR[i] = wsR[(size_t)atom * 96 + i];
  __syncthreads();

  // dR[f][c] = sum_g dF~[f,g] R[g,c] + (f<M2) sum_f' dF~[f',f] R[f',c]
  for (int q = 0; q < 3; ++q) {
    int j = lane + q * 32;
    int f = j >> 2, c = j & 3;
    float acc = 0.0f;
#pragma unroll
    for (int g = 0; g < M2_; ++g) acc += sDF[f * M2_ + g] * sR[g * 4 + c];
    if (f < M2_) {
#pragma unroll
      for (int fp = 0; fp < M1_; ++fp) acc += sDF[fp * M2_ + f] * sR[fp * 4 + c];
    }
    sDR[j] = acc;
  }
  __syncthreads();

  const float invspan = 1.0f / (RC_ - RMIN_);
  const float invMN = 1.0f / MN_;
  float cfx = 0.0f, cfy = 0.0f, cfz = 0.0f;
  float vir[9];
#pragma unroll
  for (int i = 0; i < 9; ++i) vir[i] = 0.0f;

  for (int it = 0; it < K_ / 32; ++it) {
    int k = it * 32 + lane;
    const float* p = ImageDR + ((size_t)atom * K_ + k) * 4;
    float x = p[1], y = p[2], z = p[3];
    int ln = list_neigh[(size_t)atom * K_ + k];
    float valid = (ln >= 0) ? 1.0f : 0.0f;
    float r = sqrtf(x * x + y * y + z * z + 1e-12f);
    float u = fminf(fmaxf((r - RMIN_) * invspan, 0.0f), 1.0f);
    float fc = 0.5f * (cosf(PI_ * u) + 1.0f) * valid;
    float xc = 2.0f * (r - RMIN_) * invspan - 1.0f;
    float T[BETA_], Tp[BETA_];
    T[0] = 1.0f;
    T[1] = xc;
    Tp[0] = 0.0f;
    Tp[1] = 1.0f;
#pragma unroll
    for (int j = 2; j < BETA_; ++j) {
      T[j] = 2.0f * xc * T[j - 1] - T[j - 2];
      Tp[j] = 2.0f * T[j - 1] + 2.0f * xc * Tp[j - 1] - Tp[j - 2];
    }
    float invr = valid / r;
    float sr1 = x * invr, sr2 = y * invr, sr3 = z * invr;
    int tk = k >> 7;
    const float* cc = &sC[tk * M1_ * BETA_];

    float dfc_acc = 0.0f, dx_acc = 0.0f;
    float dsr1 = 0.0f, dsr2 = 0.0f, dsr3 = 0.0f;
#pragma unroll
    for (int f = 0; f < M1_; ++f) {
      float sraw = 0.0f, sder = 0.0f;
#pragma unroll
      for (int j = 0; j < BETA_; ++j) {
        float cfj = cc[f * BETA_ + j];
        sraw = fmaf(cfj, T[j], sraw);
        sder = fmaf(cfj, Tp[j], sder);
      }
      float sval = sraw * fc;
      float dsf = invMN * (sDR[f * 4 + 0] * valid + sDR[f * 4 + 1] * sr1 +
                           sDR[f * 4 + 2] * sr2 + sDR[f * 4 + 3] * sr3);
      dfc_acc += dsf * sraw;
      dx_acc += dsf * fc * sder;
      dsr1 += sDR[f * 4 + 1] * sval;
      dsr2 += sDR[f * 4 + 2] * sval;
      dsr3 += sDR[f * 4 + 3] * sval;
    }
    dsr1 *= invMN;
    dsr2 *= invMN;
    dsr3 *= invMN;

    float dfcdr = -0.5f * PI_ * sinf(PI_ * u) * valid * invspan;
    float drtot = dx_acc * 2.0f * invspan + dfc_acc * dfcdr;
    float rr = 1.0f / r;
    float dot = dsr1 * x + dsr2 * y + dsr3 * z;
    float gx = drtot * x * rr + valid * (dsr1 * rr - dot * x * rr * rr * rr);
    float gy = drtot * y * rr + valid * (dsr2 * rr - dot * y * rr * rr * rr);
    float gz = drtot * z * rr + valid * (dsr3 * rr - dot * z * rr * rr * rr);

    if (ln >= 0) {
      float* fdst = outForce + ((size_t)b * N_ + ln) * 3;
      atomicAdd(&fdst[0], -gx);
      atomicAdd(&fdst[1], -gy);
      atomicAdd(&fdst[2], -gz);
    }
    cfx += gx;
    cfy += gy;
    cfz += gz;
    vir[0] += x * gx;
    vir[1] += x * gy;
    vir[2] += x * gz;
    vir[3] += y * gx;
    vir[4] += y * gy;
    vir[5] += y * gz;
    vir[6] += z * gx;
    vir[7] += z * gy;
    vir[8] += z * gz;
  }

  // wave32 reduction
  for (int off = 16; off > 0; off >>= 1) {
    cfx += __shfl_xor(cfx, off, 32);
    cfy += __shfl_xor(cfy, off, 32);
    cfz += __shfl_xor(cfz, off, 32);
#pragma unroll
    for (int i = 0; i < 9; ++i) vir[i] += __shfl_xor(vir[i], off, 32);
  }
  if (lane == 0) {
    float* fdst = outForce + ((size_t)b * N_ + n) * 3;
    atomicAdd(&fdst[0], cfx);
    atomicAdd(&fdst[1], cfy);
    atomicAdd(&fdst[2], cfz);
#pragma unroll
    for (int i = 0; i < 9; ++i) atomicAdd(&outVirial[b * 9 + i], -vir[i]);
  }
}

// ---------------------------------------------------------------------------
// Launch
// ---------------------------------------------------------------------------
extern "C" void kernel_launch(void* const* d_in, const int* in_sizes, int n_in,
                              void* d_out, int out_size, void* d_ws, size_t ws_size,
                              hipStream_t stream) {
  const int* list_neigh = (const int*)d_in[0];
  const int* ti_map = (const int*)d_in[1];
  const float* ImageDR = (const float*)d_in[3];
  const float* c_param = (const float*)d_in[6];
  const float* scale = (const float*)d_in[7];
  const float* W0 = (const float*)d_in[8];
  const float* b0 = (const float*)d_in[9];
  const float* W1 = (const float*)d_in[10];
  const float* b1 = (const float*)d_in[11];
  const float* W2 = (const float*)d_in[12];
  const float* b2 = (const float*)d_in[13];
  const float* ener_shift = (const float*)d_in[14];

  float* out = (float*)d_out;
  float* outEtot = out;                                 // 4
  float* outEi = out + B_;                              // 4096
  float* outForce = out + B_ + B_ * N_;                 // 12288
  float* outVirial = out + B_ + B_ * N_ + B_ * N_ * 3;  // 36

  float* ws = (float*)d_ws;
  float* wsFeat = ws;                                 // 4096*144 f32
  float* wsR = wsFeat + (size_t)B_ * N_ * NFEAT_;     // 4096*96  f32
  float* wsDfeat = wsR + (size_t)B_ * N_ * 96;        // 4096*144 f32
  __bf16* wsW0T = (__bf16*)(wsDfeat + (size_t)B_ * N_ * NFEAT_);  // NT*128*160
  __bf16* wsW1T = wsW0T + N_W0T;                                  // NT*128*128
  __bf16* wsW0b = wsW1T + N_W1T;                                  // NT*144*128
  __bf16* wsW1b = wsW0b + N_W0B;                                  // NT*128*128

  int outN = B_ + B_ * N_ + B_ * N_ * 3 + B_ * 9;
  zero_out_k<<<(outN + 255) / 256, 256, 0, stream>>>(out, outN);

  int cvtN = N_W0T + N_W1T + N_W0B + N_W1B;
  cvt_weights_k<<<(cvtN + 255) / 256, 256, 0, stream>>>(W0, W1, wsW0T, wsW1T, wsW0b,
                                                        wsW1b);

  feat_fwd_k<<<B_ * N_, 32, 0, stream>>>(ImageDR, list_neigh, ti_map, c_param, scale,
                                         wsFeat, wsR);

  mlp_wmma_k<<<B_ * N_ / 16, 256, 0, stream>>>(wsFeat, wsDfeat, ti_map, wsW0T, wsW1T,
                                               wsW0b, wsW1b, b0, b1, W2, b2,
                                               ener_shift, outEtot, outEi);

  feat_bwd_k<<<B_ * N_, 32, 0, stream>>>(ImageDR, list_neigh, ti_map, c_param, scale,
                                         wsR, wsDfeat, outForce, outVirial);
}
